// manual_MHA_61615600828751
// MI455X (gfx1250) — matrix operand order repaired and checked by reading, not compiled, not run on hardware
//
#include <hip/hip_runtime.h>
#include <hip/hip_bf16.h>
#include <math.h>

// ---------------------------------------------------------------------------
// MHA with global-scalar "softmax" for MI455X (gfx1250), wave32 + WMMA f16.
//
//   Q/K/V = E @ Wq/Wk/Wv           (per head, f16 operands, f32 accum)
//   O_un  = exp(QK^T/8) @ V        (flash-style, scores never hit memory;
//                                   K tile: async global->LDS (ASYNCcnt),
//                                   V tile: TDM tensor_load_to_lds (TENSORcnt))
//   total = sum over ALL e         (atomicAdd f32)
//   out   = (O_un/total) @ Wout
// ---------------------------------------------------------------------------

typedef __attribute__((ext_vector_type(16))) _Float16     v16h;
typedef __attribute__((ext_vector_type(8)))  float        v8f;
typedef __attribute__((ext_vector_type(4)))  int          v4i;
typedef __attribute__((ext_vector_type(8)))  int          v8i;
typedef __attribute__((ext_vector_type(4)))  unsigned int v4u;

constexpr int S_LEN  = 4096;
constexpr int DMODEL = 512;
constexpr int HEADS  = 8;
constexpr int DK     = 64;
constexpr float INV_SQRT_DK = 0.125f;   // 1/sqrt(64)

// workspace layout (bytes)
constexpr size_t OFF_Q    = 0;            // f16 [H][S][DK]  4 MiB
constexpr size_t OFF_K    = 4u  << 20;    // f16 [H][S][DK]  4 MiB
constexpr size_t OFF_VT   = 8u  << 20;    // f16 [H][DK][S]  4 MiB (transposed)
constexpr size_t OFF_OS   = 12u << 20;    // f32 [S][H*DK]   8 MiB (unnormalized)
constexpr size_t OFF_ESUM = 20u << 20;    // f32 scalar

#define HAS_ASYNC_LDS __has_builtin(__builtin_amdgcn_global_load_async_to_lds_b128)
#define HAS_TDM       __has_builtin(__builtin_amdgcn_tensor_load_to_lds)

__device__ __forceinline__ v8f wmma_f16(v16h a, v16h b, v8f c) {
    return __builtin_amdgcn_wmma_f32_16x16x32_f16(false, a, false, b,
                                                  (short)0, c, false, false);
}

// 16-byte global -> LDS copy; async (ASYNCcnt-tracked) when available.
__device__ __forceinline__ void copy16_g2l(const void* g, void* l) {
#if HAS_ASYNC_LDS
    __builtin_amdgcn_global_load_async_to_lds_b128(
        (__attribute__((address_space(1))) v4i*)g,
        (__attribute__((address_space(3))) v4i*)l, 0, 0);
#else
    *(int4*)l = *(const int4*)g;
#endif
}

template <int N>
__device__ __forceinline__ void wait_asynccnt() {
#if HAS_ASYNC_LDS
#if __has_builtin(__builtin_amdgcn_s_wait_asynccnt)
    __builtin_amdgcn_s_wait_asynccnt((unsigned short)N);
#else
    asm volatile("s_wait_asynccnt %0" ::"i"(N));
#endif
#endif
}

template <int N>
__device__ __forceinline__ void wait_tensorcnt() {
#if HAS_TDM
#if __has_builtin(__builtin_amdgcn_s_wait_tensorcnt)
    __builtin_amdgcn_s_wait_tensorcnt((unsigned short)N);
#else
    asm volatile("s_wait_tensorcnt %0" ::"i"(N));
#endif
#endif
}

// TDM: load a 2D tile (tile_dim0=32 elems of 2B contiguous, tile_dim1=64 rows,
// row stride = S_LEN elems) from global into LDS (linear, 64B per row).
// D# bitfields per CDNA5 ISA.
// This toolchain exposes the 6-arg builtin form:
//   (uint32x4 g0, int32x8 g1, int32x4, int32x4, int32x8, i32 cpol)
__device__ __forceinline__ void tdm_load_v_tile(const _Float16* gsrc, void* ldst) {
#if HAS_TDM
    const unsigned ldsoff =
        (unsigned)(size_t)(__attribute__((address_space(3))) void*)ldst;
    const unsigned long long ga = (unsigned long long)gsrc;
    const v4u g0 = {
        1u,                                                  // count=1, user D#
        ldsoff,                                              // lds_addr (bytes)
        (unsigned)(ga & 0xFFFFFFFFu),                        // global_addr[31:0]
        (unsigned)((ga >> 32) & 0x1FFFFFFu) | (2u << 30)     // addr[56:32], type=2
    };
    const v8i g1 = {
        (int)(1u << 16),      // wg_mask=0, data_size=1 (2B), no flags/pad
        (int)(32u << 16),     // atomic_bar_addr=0 | tensor_dim0[15:0]=32
        (int)(64u << 16),     // tensor_dim0[31:16]=0 | tensor_dim1[15:0]=64
        (int)(32u << 16),     // tensor_dim1[31:16]=0 | tile_dim0=32
        (int)64,              // tile_dim1=64 | tile_dim2=0
        (int)S_LEN,           // tensor_dim0_stride[31:0]=4096
        0,                    // stride0[47:32]=0 | tensor_dim1_stride[15:0]=0
        0                     // tensor_dim1_stride[47:16]=0
    };
    const v4i z4 = {0, 0, 0, 0};
    const v8i z8 = {0, 0, 0, 0, 0, 0, 0, 0};
    __builtin_amdgcn_tensor_load_to_lds(g0, g1, z4, z4, z8, 0);
#endif
}

// --- A fragment (16x32, f16), ISA 16-bit A layout -------------------------
// lane(0..15): row=lane,    K = k0+{0..7, 16..23}
// lane(16..31):row=lane-16, K = k0+{8..15,24..31}
__device__ __forceinline__ v16h load_a_f32(const float* src, int row0, int k0, int ld) {
    const int lane = threadIdx.x & 31, lo = lane & 15, hi = lane >> 4;
    const float* r = src + (size_t)(row0 + lo) * ld + k0 + hi * 8;
    v16h a;
#pragma unroll
    for (int j = 0; j < 8; ++j) {
        const int kk = (j < 4) ? (2 * j) : (16 + 2 * (j - 4));
        a[2 * j]     = (_Float16)r[kk];
        a[2 * j + 1] = (_Float16)r[kk + 1];
    }
    return a;
}

__device__ __forceinline__ v16h load_a_f32_scaled(const float* src, int row0, int k0,
                                                  int ld, float scale) {
    const int lane = threadIdx.x & 31, lo = lane & 15, hi = lane >> 4;
    const float* r = src + (size_t)(row0 + lo) * ld + k0 + hi * 8;
    v16h a;
#pragma unroll
    for (int j = 0; j < 8; ++j) {
        const int kk = (j < 4) ? (2 * j) : (16 + 2 * (j - 4));
        a[2 * j]     = (_Float16)(r[kk] * scale);
        a[2 * j + 1] = (_Float16)(r[kk + 1] * scale);
    }
    return a;
}

__device__ __forceinline__ v16h load_a_f16(const _Float16* src, int row0, int k0, int ld) {
    const int lane = threadIdx.x & 31, lo = lane & 15, hi = lane >> 4;
    const _Float16* r = src + (size_t)(row0 + lo) * ld + k0 + hi * 8;
    v16h a;
#pragma unroll
    for (int j = 0; j < 8; ++j) {
        const int kk = (j < 4) ? (2 * j) : (16 + 2 * (j - 4));
        a[2 * j]     = r[kk];
        a[2 * j + 1] = r[kk + 1];
    }
    return a;
}

// --- B fragment (32x16, f16) from n-major / k-contiguous storage ----------
// element B[k][n] at src[n*ld + k]; lane = column n, wave halves split K.
// Per-lane reads K = k0+8*hi+{0..7, 16..23} (matches A table order).
__device__ __forceinline__ v16h load_b_kcont_f16(const _Float16* src, int n0, int k0, int ld) {
    const int lane = threadIdx.x & 31, lo = lane & 15, hi = lane >> 4;
    const _Float16* r = src + (size_t)(n0 + lo) * ld + k0 + hi * 8;
    v16h b;
#pragma unroll
    for (int j = 0; j < 16; ++j) b[j] = r[j < 8 ? j : j + 8];
    return b;
}

// --- B fragment from row-major f32 W: element B[k][n] at src[k*ld + n] ----
__device__ __forceinline__ v16h load_b_rowmajor_f32(const float* src, int k0, int n0, int ld) {
    const int lane = threadIdx.x & 31, lo = lane & 15, hi = lane >> 4;
    const float* c = src + (size_t)(k0 + hi * 8) * ld + n0 + lo;
    v16h b;
#pragma unroll
    for (int j = 0; j < 16; ++j) b[j] = (_Float16)c[(size_t)(j < 8 ? j : j + 8) * ld];
    return b;
}

// ---------------------------------------------------------------------------
__global__ void init_kernel(float* esum) {
    if (threadIdx.x == 0) *esum = 0.0f;
}

// One wave computes a 16x64 tile of Q, K, or V for one head.
// grid = (S/16, HEADS, 3), block = 32
__global__ void proj_kernel(const float* __restrict__ E,
                            const float* __restrict__ Wq,
                            const float* __restrict__ Wk,
                            const float* __restrict__ Wv,
                            _Float16* __restrict__ Qh,
                            _Float16* __restrict__ Kh,
                            _Float16* __restrict__ Vth) {
    const int m0 = blockIdx.x * 16;
    const int h  = blockIdx.y;
    const int which = blockIdx.z;
    const float* W = (which == 0 ? Wq : (which == 1 ? Wk : Wv)) + (size_t)h * DMODEL * DK;

    v8f acc[4] = {};
    for (int d0 = 0; d0 < DMODEL; d0 += 32) {
        if (d0 + 32 < DMODEL)
            __builtin_prefetch(W + (size_t)(d0 + 32 + (threadIdx.x & 31)) * DK, 0, 1);
        const v16h a = load_a_f32(E, m0, d0, DMODEL);
#pragma unroll
        for (int n = 0; n < 4; ++n) {
            const v16h b = load_b_rowmajor_f32(W, d0, n * 16, DK);
            acc[n] = wmma_f16(a, b, acc[n]);
        }
    }
    const int lane = threadIdx.x & 31, lo = lane & 15, hi = lane >> 4;
    if (which == 2) {  // V stored transposed: Vt[h][v][t]
        _Float16* dst = Vth + (size_t)h * DK * S_LEN;
#pragma unroll
        for (int n = 0; n < 4; ++n)
#pragma unroll
            for (int r = 0; r < 8; ++r)
                dst[(size_t)(n * 16 + lo) * S_LEN + (m0 + r + 8 * hi)] = (_Float16)acc[n][r];
    } else {           // Q/K row-major [h][s][dk]
        _Float16* dst = (which == 0 ? Qh : Kh) + (size_t)h * S_LEN * DK;
#pragma unroll
        for (int n = 0; n < 4; ++n)
#pragma unroll
            for (int r = 0; r < 8; ++r)
                dst[(size_t)(m0 + r + 8 * hi) * DK + n * 16 + lo] = (_Float16)acc[n][r];
    }
}

// Flash-style unnormalized attention.
// Block = 128 threads (4 waves) = 64 query rows of one head.
// 32-key tiles double-buffered in LDS: K via async copies, V via TDM.
// grid = (S/64, HEADS), block = 128
__global__ void __launch_bounds__(128)
attn_kernel(const _Float16* __restrict__ Qh,
            const _Float16* __restrict__ Kh,
            const _Float16* __restrict__ Vth,
            float* __restrict__ Osum,
            float* __restrict__ esum) {
    __shared__ __align__(16) _Float16 Klds[2][32 * DK];   // [key][dk]
    __shared__ __align__(16) _Float16 Vlds[2][DK * 32];   // [v][key]
    __shared__ __align__(16) _Float16 etile[4][16 * 32];  // per-wave D->A bounce

    const int tid  = threadIdx.x;
    const int wave = tid >> 5;
    const int lane = tid & 31, lo = lane & 15, hi = lane >> 4;
    const int h    = blockIdx.y;
    const int m0   = blockIdx.x * 64 + wave * 16;

    const _Float16* Q  = Qh  + (size_t)h * S_LEN * DK;
    const _Float16* K  = Kh  + (size_t)h * S_LEN * DK;
    const _Float16* Vt = Vth + (size_t)h * DK * S_LEN;

    // Q fragments for the whole row-block (d_k = 64 -> two K=32 steps)
    const v16h qa0 = load_a_f16(Q, m0, 0,  DK);
    const v16h qa1 = load_a_f16(Q, m0, 32, DK);

    // K tile is 4KB contiguous: 2 async b128 per thread (2 per wave per tile).
    auto stage_k = [&](int buf, int t0) {
        const char* gk = (const char*)(K + (size_t)t0 * DK) + tid * 32;
        char*       lk = (char*)&Klds[buf][0] + tid * 32;
        copy16_g2l(gk, lk);
        copy16_g2l(gk + 16, lk + 16);
    };
    // V tile: 64 rows x 64B with stride S*2 B -> one TDM descriptor (wave 0),
    // or per-lane async copies if TDM builtin unavailable.
    auto stage_v = [&](int buf, int t0) {
#if HAS_TDM
        if (wave == 0) tdm_load_v_tile(Vt + t0, &Vlds[buf][0]);
#else
        const int seg = tid >> 1, part = tid & 1;
        const char* gv = (const char*)(Vt + (size_t)seg * S_LEN + t0) + part * 32;
        char*       lv = (char*)&Vlds[buf][0] + seg * 64 + part * 32;
        copy16_g2l(gv, lv);
        copy16_g2l(gv + 16, lv + 16);
#endif
    };
#if HAS_TDM
    constexpr int ASYNC_PER_TILE = 2;   // K only
#else
    constexpr int ASYNC_PER_TILE = 4;   // K + V
#endif

    stage_k(0, 0);
    stage_v(0, 0);

    v8f   acc_o[4] = {};
    float es = 0.0f;

    for (int t0 = 0; t0 < S_LEN; t0 += 32) {
        const int cur = (t0 >> 5) & 1;
        if (t0 + 32 < S_LEN) {
            stage_k(cur ^ 1, t0 + 32);       // prefetch next tile
            stage_v(cur ^ 1, t0 + 32);
            wait_asynccnt<ASYNC_PER_TILE>(); // current K tile landed
            wait_tensorcnt<1>();             // current V tile landed (wave 0)
        } else {
            wait_asynccnt<0>();
            wait_tensorcnt<0>();
        }
        __syncthreads();                     // current tile visible to all waves

        const _Float16* Kt = &Klds[cur][0];
        const _Float16* Vv = &Vlds[cur][0];

        // scores: two 16x16 tiles covering keys [t0, t0+32)
        v8f st0 = {}, st1 = {};
        st0 = wmma_f16(qa0, load_b_kcont_f16(Kt, 0,  0,  DK), st0);
        st1 = wmma_f16(qa0, load_b_kcont_f16(Kt, 16, 0,  DK), st1);
        st0 = wmma_f16(qa1, load_b_kcont_f16(Kt, 0,  32, DK), st0);
        st1 = wmma_f16(qa1, load_b_kcont_f16(Kt, 16, 32, DK), st1);

        // WAR fence: this wave's previous ds_loads from etile must drain
        asm volatile("s_wait_dscnt 0" ::: "memory");
        _Float16* et = &etile[wave][0];
#pragma unroll
        for (int r = 0; r < 8; ++r) {
            const float e0 = __expf(st0[r] * INV_SQRT_DK);
            const float e1 = __expf(st1[r] * INV_SQRT_DK);
            es += e0 + e1;
            et[(r + 8 * hi) * 32 + lo]      = (_Float16)e0;
            et[(r + 8 * hi) * 32 + 16 + lo] = (_Float16)e1;
        }
        // RAW fence before cross-lane re-read in A layout
        asm volatile("s_wait_dscnt 0" ::: "memory");
        v16h ea;
        {
            const _Float16* r = et + lo * 32 + hi * 8;
#pragma unroll
            for (int j = 0; j < 8; ++j) {
                const int kk = (j < 4) ? (2 * j) : (16 + 2 * (j - 4));
                ea[2 * j]     = r[kk];
                ea[2 * j + 1] = r[kk + 1];
            }
        }
        // O += e @ V   (K dim = 32 keys, 4 n-tiles of d_v)
#pragma unroll
        for (int n = 0; n < 4; ++n)
            acc_o[n] = wmma_f16(ea, load_b_kcont_f16(Vv, n * 16, 0, 32), acc_o[n]);

        __syncthreads();                     // all reads done before buffer reuse
    }

    // unnormalized output, concat layout [s][h*64 + v]
#pragma unroll
    for (int n = 0; n < 4; ++n)
#pragma unroll
        for (int r = 0; r < 8; ++r)
            Osum[(size_t)(m0 + r + 8 * hi) * (HEADS * DK) + h * DK + n * 16 + lo] =
                acc_o[n][r];

    // wave-reduce partial exp-sum, single atomic per wave
#pragma unroll
    for (int off = 16; off > 0; off >>= 1) es += __shfl_xor(es, off, 32);
    if (lane == 0) atomicAdd(esum, es);
}

// out = (Osum / total) @ Wout ; one wave -> 16x64 output tile
// grid = (S/16, DMODEL/64), block = 32
__global__ void out_kernel(const float* __restrict__ Osum,
                           const float* __restrict__ Wout,
                           const float* __restrict__ esum,
                           float* __restrict__ Out) {
    const int m0 = blockIdx.x * 16;
    const int n0 = blockIdx.y * 64;
    const float inv = 1.0f / *esum;

    v8f acc[4] = {};
    for (int d0 = 0; d0 < DMODEL; d0 += 32) {
        const v16h a = load_a_f32_scaled(Osum, m0, d0, HEADS * DK, inv);
#pragma unroll
        for (int n = 0; n < 4; ++n) {
            const v16h b = load_b_rowmajor_f32(Wout, d0, n0 + n * 16, DMODEL);
            acc[n] = wmma_f16(a, b, acc[n]);
        }
    }
    const int lane = threadIdx.x & 31, lo = lane & 15, hi = lane >> 4;
#pragma unroll
    for (int n = 0; n < 4; ++n)
#pragma unroll
        for (int r = 0; r < 8; ++r)
            Out[(size_t)(m0 + r + 8 * hi) * DMODEL + n0 + n * 16 + lo] = acc[n][r];
}

// ---------------------------------------------------------------------------
extern "C" void kernel_launch(void* const* d_in, const int* in_sizes, int n_in,
                              void* d_out, int out_size, void* d_ws, size_t ws_size,
                              hipStream_t stream) {
    const float* E    = (const float*)d_in[0];   // (4096, 512)
    const float* Wq   = (const float*)d_in[1];   // (8, 512, 64)
    const float* Wk   = (const float*)d_in[2];
    const float* Wv   = (const float*)d_in[3];
    const float* Wout = (const float*)d_in[4];   // (512, 512)

    char* ws = (char*)d_ws;
    _Float16* Qh  = (_Float16*)(ws + OFF_Q);
    _Float16* Kh  = (_Float16*)(ws + OFF_K);
    _Float16* Vth = (_Float16*)(ws + OFF_VT);
    float*    Os  = (float*)(ws + OFF_OS);
    float*    es  = (float*)(ws + OFF_ESUM);
    float*    Out = (float*)d_out;

    init_kernel<<<dim3(1), dim3(32), 0, stream>>>(es);
    proj_kernel<<<dim3(S_LEN / 16, HEADS, 3), dim3(32), 0, stream>>>(
        E, Wq, Wk, Wv, Qh, Kh, Vth);
    attn_kernel<<<dim3(S_LEN / 64, HEADS), dim3(128), 0, stream>>>(
        Qh, Kh, Vth, Os, es);
    out_kernel<<<dim3(S_LEN / 16, DMODEL / 64), dim3(32), 0, stream>>>(
        Os, Wout, es, Out);
}
